// S4Layer_11605001634066
// MI455X (gfx1250) — matrix operand order repaired and checked by reading, not compiled, not run on hardware
//
#include <hip/hip_runtime.h>
#include <hip/hip_bf16.h>
#include <math.h>

// S4D layer for MI455X (gfx1250, wave32).
//   B=8, H=256, L=4096, N2=32 (one wave per (b,h); one lane per state n)
// K0a: per-(h,n) discretization:  lambda = exp(dt*A), Ck = 2*C*(lambda-1)/A
// K0b: W_out (512x256 f32) -> f16
// K1 : diagonal recurrence scan (exact ZOH conv), time-blocked by 32 with a
//      per-wave LDS transpose for the state reduction; + u*D skip + exact
//      GELU -> g (f16)
// K2 : WMMA f32_16x16x32_f16 GEMM  out[b,o,l] = sum_h W[o,h]*g[b,h,l] + bias,
//      fused GLU:  d_out = a * sigmoid(gate)

#define BB 8
#define HH 256
#define LL 4096
#define NN 32
#define OO 512  // 2*H

typedef __attribute__((ext_vector_type(16))) _Float16 v16h;
typedef __attribute__((ext_vector_type(8)))  _Float16 v8h;
typedef __attribute__((ext_vector_type(8)))  float    v8f;

// ---------------------------------------------------------------- K0a: params
__global__ void s4d_params_kernel(const float* __restrict__ log_dt,
                                  const float* __restrict__ log_A_real,
                                  const float* __restrict__ A_imag,
                                  const float* __restrict__ C_re,
                                  const float* __restrict__ C_im,
                                  float* __restrict__ lamR,
                                  float* __restrict__ lamI,
                                  float* __restrict__ ckR,
                                  float* __restrict__ ckI) {
  int idx = blockIdx.x * blockDim.x + threadIdx.x;   // h*NN + n, 8192 total
  if (idx >= HH * NN) return;
  int h = idx / NN;
  float dt = expf(log_dt[h]);
  float Ar = -expf(log_A_real[idx]);
  float Ai = A_imag[idx];
  float ar = Ar * dt, ai = Ai * dt;
  float er = expf(ar);
  float lr = er * cosf(ai);
  float li = er * sinf(ai);
  // q = (lambda - 1) / A   (complex divide)
  float dr = lr - 1.0f, di = li;
  float den = Ar * Ar + Ai * Ai;
  float inv = 1.0f / den;
  float qr = (dr * Ar + di * Ai) * inv;
  float qi = (di * Ar - dr * Ai) * inv;
  float cr = C_re[idx], ci = C_im[idx];
  // Ck = 2 * C * q  (factor 2 for conjugate pairs folded in here)
  float ckr = 2.0f * (cr * qr - ci * qi);
  float cki = 2.0f * (cr * qi + ci * qr);
  lamR[idx] = lr; lamI[idx] = li; ckR[idx] = ckr; ckI[idx] = cki;
}

// ---------------------------------------------------------------- K0b: W->f16
__global__ void w_half_kernel(const float* __restrict__ W,
                              _Float16* __restrict__ Wh) {
  int i = blockIdx.x * blockDim.x + threadIdx.x;     // OO*HH = 131072
  if (i < OO * HH) Wh[i] = (_Float16)W[i];
}

// ---------------------------------------------------------------- K1: scan
// One wave32 per (b,h). Lane n holds complex state s_n. Time is blocked in
// tiles of 32 steps:
//   - u for the tile is one coalesced load, broadcast per step via readlane
//   - per-step contribution Re(Ck_n * s_n) goes to a per-wave LDS tile
//     c[t][n] (rows padded to 36 dwords: conflict-free writes, 16B-aligned
//     rows for b128 reads)
//   - lane t then row-sums c[t][*] -> y[l0+t], fuses skip + exact GELU, and
//     the wave stores 32 f16 outputs coalesced.
__global__ void s4d_scan_kernel(const float* __restrict__ u_in,
                                const float* __restrict__ lamR,
                                const float* __restrict__ lamI,
                                const float* __restrict__ ckR,
                                const float* __restrict__ ckI,
                                const float* __restrict__ Dvec,
                                _Float16* __restrict__ g) {
  __shared__ float csum[8][32][36];    // per-wave 32x36 tile, 36 KB total

  const int gtid = blockIdx.x * blockDim.x + threadIdx.x;
  const int wave = gtid >> 5;
  const int wv   = threadIdx.x >> 5;   // wave index within block (0..7)
  const int lane = threadIdx.x & 31;   // lane == state index n (N2 == 32)
  const int b = wave / HH;
  const int h = wave % HH;
  const int pidx = h * NN + lane;

  const float lr  = lamR[pidx];
  const float li  = lamI[pidx];
  const float ckr = ckR[pidx];
  const float cki = ckI[pidx];
  const float Dh  = Dvec[h];

  const float* __restrict__ u = u_in + ((size_t)(b * HH + h)) * LL;
  _Float16* __restrict__ gp   = g    + ((size_t)(b * HH + h)) * LL;

  float sr = 0.0f, si = 0.0f;
  for (int l0 = 0; l0 < LL; l0 += 32) {
    const float uv = u[l0 + lane];                    // coalesced 128B load
    #pragma unroll
    for (int t = 0; t < 32; ++t) {
      float ul = __shfl(uv, t, 32);                   // v_readlane -> sgpr
      float nsr = fmaf(lr, sr, fmaf(-li, si, ul));
      float nsi = fmaf(lr, si,  li * sr);
      sr = nsr; si = nsi;
      csum[wv][t][lane] = fmaf(ckr, sr, -cki * si);   // Re(Ck * s)
    }
    __syncthreads();
    // lane t reduces over the 32 states for output step l0 + t
    float acc = 0.0f;
    #pragma unroll
    for (int n = 0; n < 32; ++n) acc += csum[wv][lane][n];
    float y  = acc + uv * Dh;                         // skip connection
    float ge = 0.5f * y * (1.0f + erff(y * 0.70710678118654752f));
    gp[l0 + lane] = (_Float16)ge;                     // coalesced 64B store
    __syncthreads();
  }
}

// ---------------------------------------------------------------- K2: GEMM+GLU
// Grid: B * (L/16) workgroups of 256 threads (8 waves).
// LDS holds one activation tile gs[n][k] (N=16 x K=256 f16, N-major so the
// WMMA B-fragment is two 16B LDS loads per lane).
// Wave w computes output-row tiles ot = w and ot = w+8, each as a pair of
// accumulators: rows [ot*16,+16) ("a" half) and [256+ot*16,+16) ("gate").
__global__ void s4d_gemm_glu_kernel(const _Float16* __restrict__ g,
                                    const _Float16* __restrict__ Wh,
                                    const float* __restrict__ b_out,
                                    float* __restrict__ out) {
  __shared__ _Float16 gs[16][HH];      // 8 KB

  const int blk   = blockIdx.x;        // [0, B*(L/16))
  const int b     = blk >> 8;          // L/16 = 256 tiles per batch
  const int ltile = blk & 255;
  const int l0    = ltile * 16;

  // ---- stage activation tile: thread t owns channel h=t, 16 l's ----
  {
    int h = threadIdx.x;               // 256 threads == HH channels
    const _Float16* src = g + ((size_t)(b * HH + h)) * LL + l0;
    #pragma unroll
    for (int n = 0; n < 16; ++n) gs[n][h] = src[n];
  }
  __syncthreads();

  const int lane = threadIdx.x & 31;
  const int w    = threadIdx.x >> 5;   // wave id, 0..7
  const int mrow = lane & 15;          // A-fragment row (M) and B column (N)
  const int hi   = lane >> 4;          // lane-half selector

  #pragma unroll
  for (int p = 0; p < 2; ++p) {
    const int ot = w + p * 8;                         // 16 row-tiles total
    const _Float16* __restrict__ WrA = Wh + (size_t)(ot * 16 + mrow) * HH;
    const _Float16* __restrict__ WrG = Wh + (size_t)(256 + ot * 16 + mrow) * HH;

    v8f accA = {}; v8f accG = {};

    #pragma unroll
    for (int k0 = 0; k0 < HH; k0 += 32) {
      __builtin_prefetch(WrA + k0 + 64, 0, 1);        // global_prefetch_b8
      // B fragment: element e -> (K = k0 + (e&7) + 16*(e>>3) + 8*hi, N = mrow)
      v16h bfrag = __builtin_shufflevector(*(const v8h*)&gs[mrow][k0 + hi * 8],
                   *(const v8h*)&gs[mrow][k0 + 16 + hi * 8],
                   0,1,2,3,4,5,6,7,8,9,10,11,12,13,14,15);
      // A fragment: element e -> K = k0 + (e&7) + 16*(e>>3) + 8*hi, M = mrow
      v8h aloA = *(const v8h*)(WrA + k0 + hi * 8);
      v8h ahiA = *(const v8h*)(WrA + k0 + 16 + hi * 8);
      v8h aloG = *(const v8h*)(WrG + k0 + hi * 8);
      v8h ahiG = *(const v8h*)(WrG + k0 + 16 + hi * 8);
      v16h afA = __builtin_shufflevector(aloA, ahiA,
                   0,1,2,3,4,5,6,7,8,9,10,11,12,13,14,15);
      v16h afG = __builtin_shufflevector(aloG, ahiG,
                   0,1,2,3,4,5,6,7,8,9,10,11,12,13,14,15);
      accA = __builtin_amdgcn_wmma_f32_16x16x32_f16(
                false, afA, false, bfrag, (short)0, accA, false, false);
      accG = __builtin_amdgcn_wmma_f32_16x16x32_f16(
                false, afG, false, bfrag, (short)0, accG, false, false);
    }

    // ---- epilogue: bias + GLU, C/D layout: VGPR r -> M = r + 8*hi ----
    #pragma unroll
    for (int r = 0; r < 8; ++r) {
      int m  = r + 8 * hi;
      int o  = ot * 16 + m;                           // [0, 256)
      float av = accA[r] + b_out[o];
      float gv = accG[r] + b_out[o + 256];
      float ov = av * (1.0f / (1.0f + expf(-gv)));
      out[((size_t)(b * HH + o)) * LL + l0 + mrow] = ov;
    }
  }
}

// ---------------------------------------------------------------- launch
extern "C" void kernel_launch(void* const* d_in, const int* in_sizes, int n_in,
                              void* d_out, int out_size, void* d_ws, size_t ws_size,
                              hipStream_t stream) {
  (void)in_sizes; (void)n_in; (void)out_size; (void)ws_size;
  const float* hs         = (const float*)d_in[0];   // (B,H,L)
  const float* log_dt     = (const float*)d_in[1];   // (H,)
  const float* log_A_real = (const float*)d_in[2];   // (H,N2)
  const float* A_imag     = (const float*)d_in[3];   // (H,N2)
  const float* C_re       = (const float*)d_in[4];   // (H,N2)
  const float* C_im       = (const float*)d_in[5];   // (H,N2)
  const float* Dv         = (const float*)d_in[6];   // (H,)
  const float* W_out      = (const float*)d_in[7];   // (2H,H)
  const float* b_out      = (const float*)d_in[8];   // (2H,)
  float* out = (float*)d_out;                        // (B,H,L)

  // workspace layout
  float* lamR = (float*)d_ws;                        // 8192 f32
  float* lamI = lamR + HH * NN;
  float* ckR  = lamI + HH * NN;
  float* ckI  = ckR  + HH * NN;
  _Float16* Wh = (_Float16*)(ckI + HH * NN);         // 512*256 f16
  _Float16* g  = Wh + OO * HH;                       // B*H*L f16 (16 MB)

  s4d_params_kernel<<<(HH * NN + 255) / 256, 256, 0, stream>>>(
      log_dt, log_A_real, A_imag, C_re, C_im, lamR, lamI, ckR, ckI);

  w_half_kernel<<<(OO * HH + 255) / 256, 256, 0, stream>>>(W_out, Wh);

  // B*H waves, 8 waves per block -> 256 blocks
  s4d_scan_kernel<<<(BB * HH * 32) / 256, 256, 0, stream>>>(
      hs, lamR, lamI, ckR, ckI, Dv, g);

  // B * (L/16) = 2048 blocks of 8 waves
  s4d_gemm_glu_kernel<<<BB * (LL / 16), 256, 0, stream>>>(g, Wh, b_out, out);
}
